// LatentReverb_32993938767932
// MI455X (gfx1250) — matrix operand order repaired and checked by reading, not compiled, not run on hardware
//
#include <hip/hip_runtime.h>
#include <hip/hip_bf16.h>
#include <math.h>

// Geometry constants from the reference
#define B_   2
#define C_   4
#define H_   64
#define W_   64
#define HW_  (H_ * W_)          // 4096
#define CHW_ (C_ * HW_)         // 16384
#define NREFL 8
#define NHEADS 4

typedef __attribute__((ext_vector_type(16))) _Float16 v16h;
typedef __attribute__((ext_vector_type(8)))  float    v8f;

__device__ __forceinline__ float sigm(float x) { return 1.0f / (1.0f + __expf(-x)); }

// ---------------------------------------------------------------------------
// Kernel 1: fused 8-iteration reverb loop + spatial 3x3 conv.
// One workgroup per batch. LDS (dynamic, 192KB of the 320KB WGP pool):
// fb_buf, ping, pong planes of C*H*W floats each. wet accumulates in
// registers (16 elems/thread @ 1024 threads).
// ---------------------------------------------------------------------------
__global__ __launch_bounds__(1024)
void reverb_kernel(const float* __restrict__ x,
                   const float* __restrict__ refl_w,
                   const float* __restrict__ refl_d,
                   const float* __restrict__ sc_w,  const float* __restrict__ sc_b,
                   const float* __restrict__ fb_w,  const float* __restrict__ fb_b,
                   const float* __restrict__ d1_w,  const float* __restrict__ d1_b,
                   const float* __restrict__ d2_w,  const float* __restrict__ d2_b,
                   float* __restrict__ wet_out,      // [B, C, HW]
                   float* __restrict__ spatial_out)  // [B, 2C, HW]
{
    extern __shared__ float lds[];
    float* fb = lds;               // CHW_
    float* ra = lds + CHW_;        // CHW_
    float* rb = lds + 2 * CHW_;    // CHW_

    const int b = blockIdx.x;
    const int t = threadIdx.x;
    const float* xb = x + b * CHW_;

    __builtin_prefetch(xb + t, 0, 0);   // global_prefetch_b8 warm-up

    // Stage x into ra, init fb = 0.1*x
    #pragma unroll
    for (int k = 0; k < 16; ++k) {
        int idx = t + k * 1024;
        float v = xb[idx];
        ra[idx] = v;
        fb[idx] = 0.1f * v;
    }
    __syncthreads();

    // spatial = conv3x3(x, sc_w) + sc_b  -> [2C, HW]
    for (int k = 0; k < 32; ++k) {
        int o  = t + k * 1024;
        int co = o >> 12;        // 0..7
        int hw = o & (HW_ - 1);
        int h  = hw >> 6, w = hw & 63;
        float acc = sc_b[co];
        #pragma unroll
        for (int ci = 0; ci < C_; ++ci)
            #pragma unroll
            for (int dy = -1; dy <= 1; ++dy) {
                int hh = h + dy;
                if (hh < 0 || hh >= H_) continue;
                #pragma unroll
                for (int dx = -1; dx <= 1; ++dx) {
                    int ww = w + dx;
                    if (ww < 0 || ww >= W_) continue;
                    acc += sc_w[((co * C_ + ci) * 3 + (dy + 1)) * 3 + (dx + 1)]
                         * ra[ci * HW_ + hh * W_ + ww];
                }
            }
        spatial_out[b * 2 * CHW_ + o] = acc;
    }
    __syncthreads();   // done reading ra; iterations may overwrite it

    float wet[16];
    #pragma unroll
    for (int k = 0; k < 16; ++k) wet[k] = 0.0f;

    float p8 = 1.0f;
    for (int i = 0; i < NREFL; ++i) {
        // 7-tap delay kernel for this reflection (computed uniformly per thread)
        float kern[7];
        {
            int kidx = (int)floorf(refl_d[i] * 0.5f);   // ROOM_SIZE = 0.5
            kidx = kidx < 0 ? 0 : (kidx > 3 ? 3 : kidx);
            #pragma unroll
            for (int u = 0; u < 7; ++u) kern[u] = 0.0f;
            if (kidx == 0) {
                kern[3] = 1.0f;
            } else {
                int ksz = 2 * kidx + 1;
                float g[7], s = 0.0f;
                for (int u = 0; u < ksz; ++u) {
                    float xx = -2.0f + 4.0f * (float)u / (float)(ksz - 1);
                    g[u] = __expf(-xx * xx);
                    s += g[u];
                }
                for (int u = 0; u < ksz; ++u) kern[3 - ksz / 2 + u] = g[u] / s;
            }
        }
        const float wi = sigm(refl_w[i]) * p8;
        p8 *= 0.8f;

        // step 1: vertical 7-tap delay line: ra = delay(fb)
        #pragma unroll
        for (int k = 0; k < 16; ++k) {
            int idx = t + k * 1024;
            int c = idx >> 12, hw = idx & (HW_ - 1);
            int h = hw >> 6, w = hw & 63;
            float s = 0.0f;
            #pragma unroll
            for (int d = -3; d <= 3; ++d) {
                int hh = h + d;
                if (hh >= 0 && hh < H_) s += kern[d + 3] * fb[c * HW_ + hh * W_ + w];
            }
            ra[idx] = s;
        }
        __syncthreads();

        // step 2: rb = conv3x3(ra, fb_w) + fb_b
        #pragma unroll
        for (int k = 0; k < 16; ++k) {
            int idx = t + k * 1024;
            int co = idx >> 12, hw = idx & (HW_ - 1);
            int h = hw >> 6, w = hw & 63;
            float acc = fb_b[co];
            #pragma unroll
            for (int ci = 0; ci < C_; ++ci)
                #pragma unroll
                for (int dy = -1; dy <= 1; ++dy) {
                    int hh = h + dy;
                    if (hh < 0 || hh >= H_) continue;
                    #pragma unroll
                    for (int dx = -1; dx <= 1; ++dx) {
                        int ww = w + dx;
                        if (ww < 0 || ww >= W_) continue;
                        acc += fb_w[((co * C_ + ci) * 3 + (dy + 1)) * 3 + (dx + 1)]
                             * ra[ci * HW_ + hh * W_ + ww];
                    }
                }
            rb[idx] = acc;
        }
        __syncthreads();

        // step 3+4: damp gate, wet accumulate, feedback update
        #pragma unroll
        for (int k = 0; k < 16; ++k) {
            int idx = t + k * 1024;
            int c = idx >> 12, hw = idx & (HW_ - 1);
            float r0 = rb[0 * HW_ + hw], r1 = rb[1 * HW_ + hw];
            float r2 = rb[2 * HW_ + hw], r3 = rb[3 * HW_ + hw];
            float h0 = d1_b[0] + d1_w[0] * r0 + d1_w[1] * r1 + d1_w[2] * r2 + d1_w[3] * r3;
            float h1 = d1_b[1] + d1_w[4] * r0 + d1_w[5] * r1 + d1_w[6] * r2 + d1_w[7] * r3;
            h0 = h0 * sigm(h0);              // silu
            h1 = h1 * sigm(h1);
            float dmp = sigm(d2_b[c] + d2_w[c * 2] * h0 + d2_w[c * 2 + 1] * h1);
            float val = rb[idx] * dmp;
            wet[k] += val * wi;
            fb[idx] += val * 0.04f;          // FEEDBACK * 0.1
        }
        __syncthreads();
    }

    #pragma unroll
    for (int k = 0; k < 16; ++k) wet_out[b * CHW_ + t + k * 1024] = wet[k];
}

// ---------------------------------------------------------------------------
// Kernel 2: attention, one workgroup per (batch, head). head_dim == 1 so the
// softmax row max has a closed form from kmax/kmin. Per 16x32 tile, numerator
// (P@v) and denominator (row-sum of P) are produced by one f16 WMMA with
// B = [v | 1 | 0...], accumulated in f32. Two independent accumulators break
// the WMMA->WMMA RAW chain; all lane-dependent selects are branch-free FMAs
// so EXEC stays all-ones with no saveexec regions in the hot loop.
// ---------------------------------------------------------------------------
__global__ __launch_bounds__(256)
void attn_kernel(const float* __restrict__ wet,         // [B, C, HW]
                 const float* __restrict__ in_proj_w,   // [3C, C]
                 const float* __restrict__ in_proj_b,   // [3C]
                 float* __restrict__ oattn)             // [B, HW, NHEADS]
{
    __shared__ float qs[HW_], ks_[HW_], vs[HW_];
    __shared__ float red[512];

    const int bh = blockIdx.x;
    const int b  = bh >> 2;
    const int h  = bh & 3;
    const int tid = threadIdx.x;

    float wq[C_], wk[C_], wv[C_];
    #pragma unroll
    for (int c = 0; c < C_; ++c) {
        wq[c] = in_proj_w[(h)          * C_ + c];
        wk[c] = in_proj_w[(C_ + h)     * C_ + c];
        wv[c] = in_proj_w[(2 * C_ + h) * C_ + c];
    }
    const float bq = in_proj_b[h], bk = in_proj_b[C_ + h], bv = in_proj_b[2 * C_ + h];
    const float* wp = wet + b * CHW_;

    float lmax = -3.4e38f, lmin = 3.4e38f;
    for (int j = tid; j < HW_; j += 256) {
        float x0 = wp[j], x1 = wp[HW_ + j], x2 = wp[2 * HW_ + j], x3 = wp[3 * HW_ + j];
        float qv = bq + wq[0] * x0 + wq[1] * x1 + wq[2] * x2 + wq[3] * x3;
        float kv = bk + wk[0] * x0 + wk[1] * x1 + wk[2] * x2 + wk[3] * x3;
        float vv = bv + wv[0] * x0 + wv[1] * x1 + wv[2] * x2 + wv[3] * x3;
        qs[j] = qv; ks_[j] = kv; vs[j] = vv;
        lmax = fmaxf(lmax, kv); lmin = fminf(lmin, kv);
    }
    red[tid] = lmax; red[256 + tid] = lmin;
    __syncthreads();
    for (int s = 128; s > 0; s >>= 1) {
        if (tid < s) {
            red[tid]       = fmaxf(red[tid], red[tid + s]);
            red[256 + tid] = fminf(red[256 + tid], red[256 + tid + s]);
        }
        __syncthreads();
    }
    const float kmax = red[0], kmin = red[256];

    const int wvid = tid >> 5;
    const int lane = tid & 31;
    const bool hi  = lane >= 16;
    const int  m   = lane & 15;
    const int  koff  = hi ? 8 : 0;   // A-matrix K-offset for this lane half
    const int  voff  = hi ? 16 : 0;  // unused; B reads vs at the same K as A (koff)
    const float selV = (m == 0) ? 1.0f : 0.0f;   // B column 0 carries v
    const float sel1 = (m == 1) ? 1.0f : 0.0f;   // B column 1 carries ones
    float* op = oattn + b * HW_ * NHEADS;

    for (int T = wvid; T < HW_ / 16; T += 8) {          // 256 tiles, 32 per wave
        const float qv   = qs[T * 16 + m];
        const float mrow = (qv >= 0.0f) ? qv * kmax : qv * kmin;
        v8f acc0 = {};
        v8f acc1 = {};
        for (int jb = 0; jb < HW_; jb += 64) {
            v16h a0, b0, a1, b1;
            #pragma unroll
            for (int e = 0; e < 16; ++e) {
                const int K = e + (e >= 8 ? 8 : 0) + koff;   // A 16x32 f16 layout
                a0[e] = (_Float16)__expf(qv * ks_[jb + K] - mrow);
                a1[e] = (_Float16)__expf(qv * ks_[jb + 32 + K] - mrow);
                // Unconditional broadcast loads + branch-free column select
                const float v0 = vs[jb + K];
                const float v1 = vs[jb + 32 + K];
                b0[e] = (_Float16)fmaf(selV, v0, sel1);
                b1[e] = (_Float16)fmaf(selV, v1, sel1);
            }
            acc0 = __builtin_amdgcn_wmma_f32_16x16x32_f16(
                       false, a0, false, b0, (short)0, acc0, false, false);
            acc1 = __builtin_amdgcn_wmma_f32_16x16x32_f16(
                       false, a1, false, b1, (short)0, acc1, false, false);
        }
        // C layout: vgpr r holds row (r + hi*8), column = lane&15.
        // Column 0 = numerator (lanes 0,16), column 1 = denominator (lanes 1,17).
        // One bpermute pulls d next to n; only lanes with m==0 divide+store.
        const int orow0 = T * 16 + (hi ? 8 : 0);
        #pragma unroll
        for (int r = 0; r < 8; ++r) {
            const float cr = acc0[r] + acc1[r];
            const float dn = __shfl(cr, lane + 1, 32);   // lane0<-lane1, lane16<-lane17
            if (m == 0) {
                op[(orow0 + r) * NHEADS + h] = __fdividef(cr, dn);  // rcp-based divide
            }
        }
    }
}

// ---------------------------------------------------------------------------
// Kernel 3: out_proj, concat(spatial, wet2), 1x1 output conv, dry/wet mix.
// ---------------------------------------------------------------------------
__global__ __launch_bounds__(256)
void mix_kernel(const float* __restrict__ x,
                const float* __restrict__ spatial,      // [B, 2C, HW]
                const float* __restrict__ oattn,        // [B, HW, NHEADS]
                const float* __restrict__ out_proj_w,   // [C, C]
                const float* __restrict__ out_proj_b,
                const float* __restrict__ oc_w,         // [C, 3C]
                const float* __restrict__ oc_b,
                float* __restrict__ out)                // [B, C, HW]
{
    int token = blockIdx.x * 256 + threadIdx.x;         // 0..B*HW-1
    int b = token >> 12, l = token & (HW_ - 1);

    float o[NHEADS];
    #pragma unroll
    for (int j = 0; j < NHEADS; ++j) o[j] = oattn[b * HW_ * NHEADS + l * NHEADS + j];

    float wet2[C_];
    #pragma unroll
    for (int c = 0; c < C_; ++c) {
        float a = out_proj_b[c];
        #pragma unroll
        for (int j = 0; j < C_; ++j) a += out_proj_w[c * C_ + j] * o[j];
        wet2[c] = a;
    }
    float sp[2 * C_];
    #pragma unroll
    for (int c = 0; c < 2 * C_; ++c) sp[c] = spatial[b * 2 * CHW_ + c * HW_ + l];

    #pragma unroll
    for (int c = 0; c < C_; ++c) {
        float p = oc_b[c];
        #pragma unroll
        for (int mch = 0; mch < 2 * C_; ++mch) p += oc_w[c * 3 * C_ + mch] * sp[mch];
        #pragma unroll
        for (int j = 0; j < C_; ++j) p += oc_w[c * 3 * C_ + 2 * C_ + j] * wet2[j];
        int idx = b * CHW_ + c * HW_ + l;
        out[idx] = 0.7f * x[idx] + 0.3f * p;            // WET_MIX = 0.3
    }
}

// ---------------------------------------------------------------------------
extern "C" void kernel_launch(void* const* d_in, const int* in_sizes, int n_in,
                              void* d_out, int out_size, void* d_ws, size_t ws_size,
                              hipStream_t stream) {
    const float* x          = (const float*)d_in[0];
    const float* refl_w     = (const float*)d_in[1];
    const float* refl_d     = (const float*)d_in[2];
    const float* sc_w       = (const float*)d_in[3];
    const float* sc_b       = (const float*)d_in[4];
    const float* fb_w       = (const float*)d_in[5];
    const float* fb_b       = (const float*)d_in[6];
    const float* d1_w       = (const float*)d_in[7];
    const float* d1_b       = (const float*)d_in[8];
    const float* d2_w       = (const float*)d_in[9];
    const float* d2_b       = (const float*)d_in[10];
    const float* in_proj_w  = (const float*)d_in[11];
    const float* in_proj_b  = (const float*)d_in[12];
    const float* out_proj_w = (const float*)d_in[13];
    const float* out_proj_b = (const float*)d_in[14];
    const float* oc_w       = (const float*)d_in[15];
    const float* oc_b       = (const float*)d_in[16];

    float* ws      = (float*)d_ws;
    float* wet     = ws;                        // B*CHW        = 32768 f
    float* spatial = ws + B_ * CHW_;            // B*2*CHW      = 65536 f
    float* oattn   = ws + 3 * B_ * CHW_;        // B*HW*NHEADS  = 32768 f
    // total workspace: 131072 floats = 512 KB

    const size_t lds_bytes = 3u * CHW_ * sizeof(float);   // 192 KB of the 320 KB WGP LDS
    reverb_kernel<<<B_, 1024, lds_bytes, stream>>>(
        x, refl_w, refl_d, sc_w, sc_b, fb_w, fb_b, d1_w, d1_b, d2_w, d2_b, wet, spatial);

    attn_kernel<<<B_ * NHEADS, 256, 0, stream>>>(wet, in_proj_w, in_proj_b, oattn);

    mix_kernel<<<(B_ * HW_) / 256, 256, 0, stream>>>(
        x, spatial, oattn, out_proj_w, out_proj_b, oc_w, oc_b, (float*)d_out);
}
